// GlobalMultiScaleAdaptKernelModule_19542101196952
// MI455X (gfx1250) — compile-verified, matrix operand order checked by reading, not run on hardware
//
#include <hip/hip_runtime.h>
#include <hip/hip_bf16.h>
#include <stddef.h>

// ---------------- problem constants ----------------
#define N_   16
#define C_   256
#define CM_  64
#define H_   96
#define W_   96
#define HW_  9216          // 96*96
#define PO_  128           // PLANES/2 (fuse output channels per branch)

// ---------------- WMMA / vector types ----------------
typedef __attribute__((ext_vector_type(16))) __bf16 v16bf;
typedef __attribute__((ext_vector_type(8)))  float  v8f;
typedef __attribute__((ext_vector_type(4)))  float  fx4;   // native vector for nontemporal ld

union frag32 { v16bf v; uint4 q[2]; };      // one lane's 32-byte bf16 fragment

// float -> bf16 (round to nearest even), bf16 -> float
static __device__ inline unsigned short f2bf(float f) {
    unsigned int u = __float_as_uint(f);
    u += 0x7fffu + ((u >> 16) & 1u);
    return (unsigned short)(u >> 16);
}
static __device__ inline float bf2f(unsigned short h) {
    return __uint_as_float(((unsigned int)h) << 16);
}

// Inverse of the 16x16x32 bf16 fragment K-map (ISA 7.12.2):
//   lane half (0:lanes0-15, 1:lanes16-31) and element index for K-within-32 `kk`
static __device__ inline int lanehalf_of_k(int kk) { return (kk >> 3) & 1; }
static __device__ inline int elem_of_k(int kk)     { return (kk & 7) + (((kk >> 4) & 1) << 3); }

// ---------------- stage 0: zero the pooling accumulator ----------------
__global__ __launch_bounds__(256) void k_zero(float* __restrict__ p, int n) {
    int i = blockIdx.x * 256 + threadIdx.x;
    if (i < n) p[i] = 0.0f;
}

// ---------------- stage 1: f = relu(conv_w @ x + b), fused x spatial-sum ----------------
// block = 256 threads = 8 waves; tile = 64 out-chan x 32 pixels; K = 256 (8 wmma steps)
// Operands live in LDS pre-swizzled to the wave32 fragment layout:
//   fragment(mt/nt, ks, lane) = 32 contiguous bytes -> 2x ds_load_b128 per operand.
__global__ __launch_bounds__(256) void k_stage1(
    const float* __restrict__ x, const float* __restrict__ conv_w,
    const float* __restrict__ conv_b, float* __restrict__ fout,
    float* __restrict__ xsum)
{
    __shared__ uint4 aSw[4 * 8 * 32 * 2];   // conv_w fragments [mt][ks][lane], 32 KB
    __shared__ uint4 bSw[2 * 8 * 32 * 2];   // x-tile fragments [nt][ks][lane], 16 KB
    __shared__ float cb[CM_];

    unsigned short* aS = (unsigned short*)aSw;
    unsigned short* bS = (unsigned short*)bSw;

    const int tid = threadIdx.x;
    const int n   = blockIdx.x / (HW_ / 32);
    const int p0  = (blockIdx.x % (HW_ / 32)) * 32;

    if (tid < CM_) cb[tid] = conv_b[tid];

    // conv_w -> swizzled LDS bf16 (vectorized global reads)
    const float4* w4 = (const float4*)conv_w;
    for (int q = tid; q < CM_ * C_ / 4; q += 256) {
        const float4 v = w4[q];
        const int m = (q * 4) / C_, k0 = (q * 4) % C_;
        const float vv[4] = {v.x, v.y, v.z, v.w};
        #pragma unroll
        for (int j = 0; j < 4; ++j) {
            const int k = k0 + j, kk = k & 31;
            const int fi = ((m >> 4) * 8 + (k >> 5)) * 32 + (m & 15) + (lanehalf_of_k(kk) << 4);
            aS[fi * 16 + elem_of_k(kk)] = f2bf(vv[j]);
        }
    }

    // x tile -> swizzled LDS bf16; fold spatial-mean partial sum (x read exactly once,
    // non-temporal so the streamed 151MB doesn't evict the L2-resident f intermediate)
    {
        const int c  = tid;                          // one channel row per thread
        const int kk = c & 31, ks = c >> 5;
        const int lh = lanehalf_of_k(kk) << 4;
        const int ii = elem_of_k(kk);
        const fx4* xp = (const fx4*)(x + ((size_t)n * C_ + c) * HW_ + p0);
        float s = 0.0f;
        #pragma unroll
        for (int q = 0; q < 8; ++q) {
            const fx4 v = __builtin_nontemporal_load(xp + q);
            #pragma unroll
            for (int j = 0; j < 4; ++j) {
                const int p = q * 4 + j;
                const float vj = v[j];
                s += vj;
                const int fi = ((p >> 4) * 8 + ks) * 32 + (p & 15) + lh;
                bS[fi * 16 + ii] = f2bf(vj);
            }
        }
        atomicAdd(&xsum[n * C_ + c], s);
    }
    __syncthreads();

    const int wave = tid >> 5, lane = tid & 31, lm = lane & 15;
    const int mt = wave >> 1;                   // 4 M-tiles of 16 (64 channels)
    const int nt = wave & 1;                    // 2 N-tiles of 16 (32 pixels)

    v8f acc = {};
    #pragma unroll
    for (int ks = 0; ks < 8; ++ks) {            // K = 256 = 8 * 32
        frag32 a, b;
        const uint4* ap = &aSw[((mt * 8 + ks) * 32 + lane) * 2];
        const uint4* bp = &bSw[((nt * 8 + ks) * 32 + lane) * 2];
        a.q[0] = ap[0]; a.q[1] = ap[1];         // 2x ds_load_b128
        b.q[0] = bp[0]; b.q[1] = bp[1];         // 2x ds_load_b128
        acc = __builtin_amdgcn_wmma_f32_16x16x32_bf16(
                  false, a.v, false, b.v, (short)0, acc, false, false);
    }

    // D layout: VGPR r -> row r (lanes 0-15) / r+8 (lanes 16-31); col = lane%16
    #pragma unroll
    for (int r = 0; r < 8; ++r) {
        const int m = mt * 16 + r + ((lane >> 4) << 3);
        const int p = p0 + nt * 16 + lm;
        const float v = acc[r] + cb[m];
        fout[((size_t)n * CM_ + m) * HW_ + p] = fmaxf(v, 0.0f);
    }
}

// ---------------- stage 2: g + dynamic depthwise kernel expansion ----------------
__global__ __launch_bounds__(256) void k_stage2(
    const float* __restrict__ conv_w, const float* __restrict__ conv_b,
    const float* __restrict__ ckw,  const float* __restrict__ ckb,
    const float* __restrict__ ck2w, const float* __restrict__ ck2b,
    const float* __restrict__ kw,   const float* __restrict__ kb,
    const float* __restrict__ k2w,  const float* __restrict__ k2b,
    const float* __restrict__ xsum,
    float* __restrict__ wdyn, float* __restrict__ watr)
{
    const int t = blockIdx.x * 256 + threadIdx.x;
    if (t >= N_ * CM_) return;
    const int n = t / CM_, cm = t % CM_;

    const float inv = 1.0f / (float)HW_;
    const float* wrow = conv_w + cm * C_;
    const float* xr   = xsum + n * C_;
    float acc = conv_b[cm];
    for (int c = 0; c < C_; ++c) acc += wrow[c] * (xr[c] * inv);
    const float g = acc > 0.0f ? acc : 0.0f;

    const float a1 = ckw[0],  b1 = ckb[0];
    const float a2 = ck2w[0], b2 = ck2b[0];
    float* wd = wdyn + (size_t)(n * CM_ + cm) * 25;
    float* wa = watr + (size_t)(n * CM_ + cm) * 9;
    #pragma unroll
    for (int kk = 0; kk < 25; ++kk) wd[kk] = a1 * (g * kw[kk] + kb[kk]) + b1;
    #pragma unroll
    for (int j = 0; j < 9; ++j)     wa[j]  = a2 * (g * k2w[j] + k2b[j]) + b2;
}

// ---------------- stage 3: dynamic depthwise (both branches) + WMMA fuse ----------------
// block = 256 threads = 8 waves; spatial tile 8x8; halo 2 serves both 5x5 and 3x3-dil2.
// Depthwise results are written straight into fragment-swizzled LDS for the fuse GEMM.
__global__ __launch_bounds__(256) void k_stage3(
    const float* __restrict__ fin,
    const float* __restrict__ wdyn, const float* __restrict__ watr,
    const float* __restrict__ adap_b, const float* __restrict__ atrous_b,
    const float* __restrict__ fuse_w, const float* __restrict__ fuse_b,
    float* __restrict__ out)
{
    __shared__ uint4 fwSw[8 * 2 * 32 * 2];          // fuse_w fragments [mt][ks][lane], 16 KB
    __shared__ uint4 dSw[4 * 2 * 32 * 2];           // dw-out fragments [nt][ks][lane],  8 KB
    __shared__ unsigned short patch[CM_][12][14];   // haloed f tile, bf16 (21.5 KB)
    __shared__ float kd[CM_][25];                   // 5x5 taps
    __shared__ float ka[CM_][9];                    // 3x3 taps
    __shared__ float fbias[PO_];
    __shared__ float dwb[2][CM_];

    unsigned short* fS = (unsigned short*)fwSw;
    unsigned short* dS = (unsigned short*)dSw;

    const int tid = threadIdx.x;
    const int blk = blockIdx.x;                     // N_ * (96/8)^2 = 2304
    const int n   = blk / 144;
    const int t2  = blk % 144;
    const int y0  = (t2 / 12) * 8, x0 = (t2 % 12) * 8;

    if (tid < PO_) fbias[tid] = fuse_b[tid];
    if (tid < CM_) { dwb[0][tid] = adap_b[tid]; dwb[1][tid] = atrous_b[tid]; }

    // fuse_w -> swizzled LDS bf16
    const float4* w4 = (const float4*)fuse_w;
    for (int q = tid; q < PO_ * CM_ / 4; q += 256) {
        const float4 v = w4[q];
        const int m = (q * 4) >> 6, k0 = (q * 4) & 63;
        const float vv[4] = {v.x, v.y, v.z, v.w};
        #pragma unroll
        for (int j = 0; j < 4; ++j) {
            const int k = k0 + j, kk = k & 31;
            const int fi = ((m >> 4) * 2 + (k >> 5)) * 32 + (m & 15) + (lanehalf_of_k(kk) << 4);
            fS[fi * 16 + elem_of_k(kk)] = f2bf(vv[j]);
        }
    }

    // haloed f patch (zero padded; f is L2-resident after stage 1)
    const float* fb = fin + (size_t)n * CM_ * HW_;
    for (int i = tid; i < CM_ * 144; i += 256) {
        const int c = i / 144, rem = i % 144, r = rem / 12, col = rem % 12;
        const int gy = y0 + r - 2, gx = x0 + col - 2;
        float v = 0.0f;
        if (gy >= 0 && gy < H_ && gx >= 0 && gx < W_)
            v = fb[(size_t)c * HW_ + gy * W_ + gx];
        patch[c][r][col] = f2bf(v);
    }
    for (int i = tid; i < CM_ * 25; i += 256) kd[i / 25][i % 25] = wdyn[(size_t)n * CM_ * 25 + i];
    for (int i = tid; i < CM_ * 9;  i += 256) ka[i / 9][i % 9]   = watr[(size_t)n * CM_ * 9 + i];
    __syncthreads();

    const int wave = tid >> 5, lane = tid & 31, lm = lane & 15;

    for (int br = 0; br < 2; ++br) {
        // ---- depthwise taps in fp32 VALU (64 chan x 64 pix / 256 threads) ----
        for (int item = tid; item < CM_ * 64; item += 256) {
            const int c = item >> 6, pix = item & 63;
            const int py = pix >> 3, px = pix & 7;
            float s = dwb[br][c];
            if (br == 0) {                          // 5x5, dil 1, pad 2
                #pragma unroll
                for (int i = 0; i < 5; ++i)
                    #pragma unroll
                    for (int j = 0; j < 5; ++j)
                        s += kd[c][i * 5 + j] * bf2f(patch[c][py + i][px + j]);
            } else {                                // 3x3, dil 2, pad 2
                #pragma unroll
                for (int i = 0; i < 3; ++i)
                    #pragma unroll
                    for (int j = 0; j < 3; ++j)
                        s += ka[c][i * 3 + j] * bf2f(patch[c][py + 2 * i][px + 2 * j]);
            }
            // write straight into the B-fragment swizzle: k = c, col = pix
            const int kk = c & 31, ks = c >> 5;
            const int fi = ((pix >> 4) * 2 + ks) * 32 + (pix & 15) + (lanehalf_of_k(kk) << 4);
            dS[fi * 16 + elem_of_k(kk)] = f2bf(s);
        }
        __syncthreads();

        // ---- fuse: y[128, 64pix] = fuse_w[128,64] @ dw[64,64pix] via WMMA ----
        const int mt = wave;                        // 8 waves -> 8 M-tiles = 128 rows
        #pragma unroll
        for (int nt = 0; nt < 4; ++nt) {            // 4 N-tiles of 16 pixels
            v8f acc = {};
            #pragma unroll
            for (int ks = 0; ks < 2; ++ks) {        // K = 64 = 2 * 32
                frag32 a, b;
                const uint4* ap = &fwSw[((mt * 2 + ks) * 32 + lane) * 2];
                const uint4* bp = &dSw[((nt * 2 + ks) * 32 + lane) * 2];
                a.q[0] = ap[0]; a.q[1] = ap[1];
                b.q[0] = bp[0]; b.q[1] = bp[1];
                acc = __builtin_amdgcn_wmma_f32_16x16x32_bf16(
                          false, a.v, false, b.v, (short)0, acc, false, false);
            }
            #pragma unroll
            for (int r = 0; r < 8; ++r) {
                const int m   = mt * 16 + r + ((lane >> 4) << 3);
                const int pix = nt * 16 + lm;
                const int py  = pix >> 3, px = pix & 7;
                // streamed output, never re-read -> non-temporal store
                __builtin_nontemporal_store(acc[r] + fbias[m],
                    &out[((size_t)n * 2 * PO_ + br * PO_ + m) * HW_ + (y0 + py) * W_ + (x0 + px)]);
            }
        }
        __syncthreads();                            // dSw reused by branch 2
    }
}

// ---------------- host-side launch ----------------
extern "C" void kernel_launch(void* const* d_in, const int* in_sizes, int n_in,
                              void* d_out, int out_size, void* d_ws, size_t ws_size,
                              hipStream_t stream) {
    (void)in_sizes; (void)n_in; (void)out_size; (void)ws_size;

    const float* x        = (const float*)d_in[0];
    const float* conv_w   = (const float*)d_in[1];
    const float* conv_b   = (const float*)d_in[2];
    const float* conv_k_w = (const float*)d_in[3];
    const float* conv_k_b = (const float*)d_in[4];
    const float* conv_k2w = (const float*)d_in[5];
    const float* conv_k2b = (const float*)d_in[6];
    const float* kern_w   = (const float*)d_in[7];
    const float* kern_b   = (const float*)d_in[8];
    const float* kern2_w  = (const float*)d_in[9];
    const float* kern2_b  = (const float*)d_in[10];
    const float* fuse_w   = (const float*)d_in[11];
    const float* fuse_b   = (const float*)d_in[12];
    const float* adap_b   = (const float*)d_in[13];
    const float* atrous_b = (const float*)d_in[14];
    float* out = (float*)d_out;

    // workspace layout
    char* ws = (char*)d_ws;
    float* xsum = (float*)ws;                                   // 16*256 f32
    float* f    = (float*)(ws + 16384);                         // 16*64*9216 f32 (36 MB)
    float* wdyn = (float*)(ws + 16384 + (size_t)N_ * CM_ * HW_ * 4);
    float* watr = wdyn + (size_t)N_ * CM_ * 25;

    k_zero<<<(N_ * C_ + 255) / 256, 256, 0, stream>>>(xsum, N_ * C_);

    k_stage1<<<N_ * (HW_ / 32), 256, 0, stream>>>(x, conv_w, conv_b, f, xsum);

    k_stage2<<<(N_ * CM_ + 255) / 256, 256, 0, stream>>>(
        conv_w, conv_b, conv_k_w, conv_k_b, conv_k2w, conv_k2b,
        kern_w, kern_b, kern2_w, kern2_b, xsum, wdyn, watr);

    k_stage3<<<N_ * 144, 256, 0, stream>>>(
        f, wdyn, watr, adap_b, atrous_b, fuse_w, fuse_b, out);
}
